// Custom_GroupedQueryAttention_13228499272236
// MI455X (gfx1250) — matrix operand order repaired and checked by reading, not compiled, not run on hardware
//
#include <hip/hip_runtime.h>
#include <hip/hip_bf16.h>
#include <stdint.h>

#define S_SEQ 2048
#define D_IN  4096
#define NH    32
#define NG    8
#define DHEAD 128
#define HDTOT (NH * DHEAD)   // 4096
#define GDTOT (NG * DHEAD)   // 1024

typedef __bf16 v16bf __attribute__((ext_vector_type(16)));
typedef float  v8f   __attribute__((ext_vector_type(8)));
typedef __bf16 bf4   __attribute__((ext_vector_type(4)));
typedef unsigned int u32x4 __attribute__((ext_vector_type(4)));
typedef int i32x4 __attribute__((ext_vector_type(4)));
typedef int i32x8 __attribute__((ext_vector_type(8)));

union Frag16 { v16bf v; uint4 u[2]; };

#if __has_builtin(__builtin_amdgcn_tensor_load_to_lds) && \
    __has_builtin(__builtin_amdgcn_s_wait_tensorcnt)
#define HAVE_TDM 1
#else
#define HAVE_TDM 0
#endif

static __device__ __forceinline__ v8f wmma_bf16(v16bf a, v16bf b, v8f c) {
  // D = A(16x32 bf16) * B(32x16 bf16) + C(16x16 f32)
  return __builtin_amdgcn_wmma_f32_16x16x32_bf16(false, a, false, b, (short)0, c,
                                                 false, false);
}

#if HAVE_TDM
// Issue a TDM 2D tile load: tileY rows x tileX bf16 elems, row stride rowElems,
// packed contiguously into LDS at lds_ptr.  D# built per CDNA5 ISA 8.3-8.5.
static __device__ __forceinline__ void tdm_load_2d(const void* gaddr, void* lds_ptr,
                                                   int rowElems, int tileX, int tileY) {
  uint64_t ga = (uint64_t)gaddr;
  uint32_t ldso = (uint32_t)(uintptr_t)lds_ptr;  // low 32 bits = LDS byte offset
  u32x4 g0;
  g0[0] = 1u;                                            // count=1, user mode
  g0[1] = ldso;                                          // lds_addr
  g0[2] = (uint32_t)ga;                                  // global_addr[31:0]
  g0[3] = (uint32_t)((ga >> 32) & 0x1FFFFFFu) | (2u << 30);  // addr[56:32]|type=2
  i32x8 g1;
  g1[0] = 0x00010000;                                    // wg_mask=0, data_size=2B
  g1[1] = (int)((rowElems & 0xFFFF) << 16);              // tensor_dim0 lo @bit48
  g1[2] = (int)(((rowElems >> 16) & 0xFFFF) |            // tensor_dim0 hi
                ((tileY & 0xFFFF) << 16));               // tensor_dim1 lo
  g1[3] = (int)(((tileY >> 16) & 0xFFFF) |               // tensor_dim1 hi
                ((tileX & 0xFFFF) << 16));               // tile_dim0
  g1[4] = (int)(tileY & 0xFFFF);                         // tile_dim1 (tile_dim2=0)
  g1[5] = (int)rowElems;                                 // tensor_dim0_stride lo
  g1[6] = 0;                                             // stride hi / dim1_stride
  g1[7] = 0;
  i32x4 z4 = {0, 0, 0, 0};
#if __has_include(<hip/amd_detail/amd_gfx1250_TDM.h>)
  i32x8 z8 = {0, 0, 0, 0, 0, 0, 0, 0};
  __builtin_amdgcn_tensor_load_to_lds(g0, g1, z4, z4, z8, 0);
#else
  __builtin_amdgcn_tensor_load_to_lds(g0, g1, z4, z4, 0);
#endif
}
#endif

// ---------------------------------------------------------------------------
// fp32 -> bf16 straight conversion (vectorized, 4 elems/thread)
// ---------------------------------------------------------------------------
__global__ void k_cvt_f32_bf16(const float* __restrict__ in,
                               __bf16* __restrict__ out, int n4) {
  int i = blockIdx.x * blockDim.x + threadIdx.x;
  if (i >= n4) return;
  float4 f = ((const float4*)in)[i];
  bf4 b;
  b.x = (__bf16)f.x; b.y = (__bf16)f.y; b.z = (__bf16)f.z; b.w = (__bf16)f.w;
  ((bf4*)out)[i] = b;
}

// ---------------------------------------------------------------------------
// in[K][N] fp32  ->  out[N][K] bf16 (32x32 LDS tile transpose)
// ---------------------------------------------------------------------------
__global__ void k_transpose_cvt(const float* __restrict__ in,
                                __bf16* __restrict__ out, int K, int N) {
  __shared__ float tile[32][33];
  int nb = N >> 5;
  int bx = blockIdx.x % nb;
  int by = blockIdx.x / nb;
  int tx = threadIdx.x & 31;
  int ty = threadIdx.x >> 5;
#pragma unroll
  for (int r = 0; r < 4; ++r)
    tile[ty + r * 8][tx] = in[(size_t)(by * 32 + ty + r * 8) * N + bx * 32 + tx];
  __syncthreads();
#pragma unroll
  for (int r = 0; r < 4; ++r)
    out[(size_t)(bx * 32 + ty + r * 8) * K + by * 32 + tx] =
        (__bf16)tile[tx][ty + r * 8];
}

// ---------------------------------------------------------------------------
// GEMM: C[S x (nStrips*128)] = A[S x Kd] * Bt^T   (Bt stored [N][Kd] bf16)
// Block = 8 waves, all sharing one 128-col strip; per K-step the B tile
// (128x32 bf16, 8 KB) is staged into LDS by the Tensor Data Mover and each
// wave builds its B fragments from LDS.  One wave -> 16 rows x 128 cols.
// MODE 0: RoPE epilogue, bf16 out, head-major [nstrip][S][128]   (Q, K)
// MODE 2: plain epilogue,  f32 out, row-major [S][nStrips*128]   (final)
// MODE 3: plain epilogue, bf16 out, transposed [nStrips*128][S]  (V)
// ---------------------------------------------------------------------------
template <int MODE>
__global__ __launch_bounds__(256, 1)
void k_gemm(const __bf16* __restrict__ A, const __bf16* __restrict__ Bt,
            const float* __restrict__ cosT, const float* __restrict__ sinT,
            void* __restrict__ outp, int Kd, int nStrips) {
  const int lane = threadIdx.x & 31;
  const int wIn  = threadIdx.x >> 5;
  const int wid  = blockIdx.x * 8 + wIn;
  const int mstrip = wid & 127;   // S/16 == 128 row-strips
  const int nstrip = wid >> 7;    // same for all 8 waves of a block
  const int m0   = mstrip << 4;
  const int half = lane >> 4;
  const int nn   = lane & 15;

  const __bf16* arow  = A + (size_t)(m0 + nn) * Kd;
  const size_t  bbase = (size_t)(nstrip * 128) * Kd;

  v8f acc[8] = {};

#if HAVE_TDM
  __shared__ __align__(128) __bf16 ldsB[128][32];   // one B tile per K-step
  for (int kk = 0; kk < Kd; kk += 32) {
    if (wIn == 0) {
      tdm_load_2d(Bt + bbase + kk, &ldsB[0][0], Kd, 32, 128);
      __builtin_amdgcn_s_wait_tensorcnt(0);
    }
    __syncthreads();
    Frag16 af;  // A 16x32: lanes=M, halves split K 0-15 / 16-31
    af.u[0] = *(const uint4*)(arow + kk + half * 16);
    af.u[1] = *(const uint4*)(arow + kk + half * 16 + 8);
    __builtin_prefetch(arow + kk + 128, 0, 0);
#pragma unroll
    for (int j = 0; j < 8; ++j) {
      Frag16 bf_;  // B 32x16: lane=N, halves split K 0-15 / 16-31
      const __bf16* bp = &ldsB[j * 16 + nn][half * 16];
      bf_.u[0] = ((const uint4*)bp)[0];
      bf_.u[1] = ((const uint4*)bp)[1];
      acc[j] = wmma_bf16(af.v, bf_.v, acc[j]);
    }
    __syncthreads();   // protect LDS tile from next iteration's DMA
  }
#else
  for (int kk = 0; kk < Kd; kk += 32) {
    Frag16 af;
    af.u[0] = *(const uint4*)(arow + kk + half * 16);
    af.u[1] = *(const uint4*)(arow + kk + half * 16 + 8);
    __builtin_prefetch(arow + kk + 128, 0, 0);
#pragma unroll
    for (int j = 0; j < 8; ++j) {
      const __bf16* bp = Bt + bbase + (size_t)(j * 16 + nn) * Kd + kk + half * 16;
      Frag16 bf_;
      bf_.u[0] = ((const uint4*)bp)[0];
      bf_.u[1] = ((const uint4*)bp)[1];
      acc[j] = wmma_bf16(af.v, bf_.v, acc[j]);
    }
  }
#endif

  if constexpr (MODE == 0) {
    __bf16* ob = (__bf16*)outp;  // [head][S][128]
#pragma unroll
    for (int j = 0; j < 8; ++j) {
#pragma unroll
      for (int i = 0; i < 8; ++i) {
        int sg = m0 + i + 8 * half;
        int d  = j * 16 + nn;
        float c  = cosT[(size_t)sg * DHEAD + d];
        float sn = sinT[(size_t)sg * DHEAD + d];
        float q  = acc[j][i];
        float qr = acc[j ^ 4][i];        // rotate_half partner (d +/- 64)
        float val = q * c + ((j >= 4) ? qr : -qr) * sn;
        ob[((size_t)nstrip * S_SEQ + sg) * DHEAD + d] = (__bf16)val;
      }
    }
  } else if constexpr (MODE == 2) {
    float* of = (float*)outp;
    const int Ntot = nStrips * 128;
#pragma unroll
    for (int j = 0; j < 8; ++j)
#pragma unroll
      for (int i = 0; i < 8; ++i) {
        int sg = m0 + i + 8 * half;
        of[(size_t)sg * Ntot + nstrip * 128 + j * 16 + nn] = acc[j][i];
      }
  } else {  // MODE 3: transposed bf16 [col][S] for V
    __bf16* ob = (__bf16*)outp;
#pragma unroll
    for (int j = 0; j < 8; ++j)
#pragma unroll
      for (int i = 0; i < 8; ++i) {
        int sg = m0 + i + 8 * half;
        ob[(size_t)(nstrip * 128 + j * 16 + nn) * S_SEQ + sg] = (__bf16)acc[j][i];
      }
  }
}

// ---------------------------------------------------------------------------
// Flash attention: one wave = 16 queries of one head, online softmax over
// causal keys in steps of 32.  q:[H][S][128]  k:[G][S][128]  v:[G*128][S]
// Per-wave causal trip counts -> no block barriers; P transposed via per-wave
// LDS tile (in-order DS).  ctx out: [S][H*128] bf16.
// ---------------------------------------------------------------------------
__global__ __launch_bounds__(256, 1)
void k_attn(const __bf16* __restrict__ qb, const __bf16* __restrict__ kb,
            const __bf16* __restrict__ vt, __bf16* __restrict__ ctx) {
  __shared__ __align__(16) __bf16 psh[8][16][32];
  const int lane = threadIdx.x & 31;
  const int wIn  = threadIdx.x >> 5;
  const int qblk = blockIdx.x & 15;   // S/128 == 16
  const int h    = blockIdx.x >> 4;
  const int g    = h >> 2;            // GROUP = 4
  const int q0   = qblk * 128 + wIn * 16;
  const int half = lane >> 4;
  const int nn   = lane & 15;
  const float scale = 0.08838834764831845f;  // 1/sqrt(128)

  Frag16 aQ[4];
  const __bf16* qrow = qb + ((size_t)h * S_SEQ + q0 + nn) * DHEAD;
#pragma unroll
  for (int c = 0; c < 4; ++c) {
    aQ[c].u[0] = *(const uint4*)(qrow + c * 32 + half * 16);
    aQ[c].u[1] = *(const uint4*)(qrow + c * 32 + half * 16 + 8);
  }

  v8f O[8] = {};
  float rmax[8], rsum[8];
#pragma unroll
  for (int i = 0; i < 8; ++i) { rmax[i] = -3.0e38f; rsum[i] = 0.0f; }

  const __bf16* kgbase = kb + (size_t)g * S_SEQ * DHEAD;
  const __bf16* vgbase = vt + (size_t)g * DHEAD * S_SEQ;

  for (int kt = 0; kt <= q0 + 15; kt += 32) {
    v8f s0 = {}, s1 = {};
#pragma unroll
    for (int c = 0; c < 4; ++c) {
      const __bf16* kp0 = kgbase + (size_t)(kt + nn) * DHEAD + c * 32 + half * 16;
      const __bf16* kp1 = kgbase + (size_t)(kt + 16 + nn) * DHEAD + c * 32 + half * 16;
      Frag16 b0, b1;
      b0.u[0] = ((const uint4*)kp0)[0]; b0.u[1] = ((const uint4*)kp0)[1];
      b1.u[0] = ((const uint4*)kp1)[0]; b1.u[1] = ((const uint4*)kp1)[1];
      s0 = wmma_bf16(aQ[c].v, b0.v, s0);
      s1 = wmma_bf16(aQ[c].v, b1.v, s1);
    }
#pragma unroll
    for (int i = 0; i < 8; ++i) {
      int srow = q0 + i + 8 * half;
      float v0 = s0[i] * scale; if (kt + nn > srow)       v0 = -3.0e38f;
      float v1 = s1[i] * scale; if (kt + 16 + nn > srow)  v1 = -3.0e38f;
      float mx = fmaxf(v0, v1);
#pragma unroll
      for (int d = 1; d < 16; d <<= 1) mx = fmaxf(mx, __shfl_xor(mx, d, 16));
      float nm = fmaxf(rmax[i], mx);
      float f  = __expf(rmax[i] - nm);
      rmax[i] = nm;
      float p0 = __expf(v0 - nm);
      float p1 = __expf(v1 - nm);
      float ps = p0 + p1;
#pragma unroll
      for (int d = 1; d < 16; d <<= 1) ps += __shfl_xor(ps, d, 16);
      rsum[i] = rsum[i] * f + ps;
#pragma unroll
      for (int j = 0; j < 8; ++j) O[j][i] *= f;
      psh[wIn][i + 8 * half][nn]      = (__bf16)p0;
      psh[wIn][i + 8 * half][16 + nn] = (__bf16)p1;
    }
    Frag16 pf;
    pf.u[0] = *(const uint4*)&psh[wIn][nn][half * 16];
    pf.u[1] = *(const uint4*)&psh[wIn][nn][half * 16 + 8];
#pragma unroll
    for (int j = 0; j < 8; ++j) {
      const __bf16* vp = vgbase + (size_t)(j * 16 + nn) * S_SEQ + kt + half * 16;
      Frag16 bv;
      bv.u[0] = ((const uint4*)vp)[0]; bv.u[1] = ((const uint4*)vp)[1];
      O[j] = wmma_bf16(pf.v, bv.v, O[j]);
    }
  }

#pragma unroll
  for (int j = 0; j < 8; ++j)
#pragma unroll
    for (int i = 0; i < 8; ++i) {
      int sg = q0 + i + 8 * half;
      float val = O[j][i] / rsum[i];
      ctx[(size_t)sg * HDTOT + h * DHEAD + j * 16 + nn] = (__bf16)val;
    }
}

// ---------------------------------------------------------------------------
extern "C" void kernel_launch(void* const* d_in, const int* in_sizes, int n_in,
                              void* d_out, int out_size, void* d_ws, size_t ws_size,
                              hipStream_t stream) {
  (void)in_sizes; (void)n_in; (void)out_size; (void)ws_size;
  const float* x    = (const float*)d_in[0];
  /* d_in[1] = causal mask, structure known statically -> unused */
  const float* cosT = (const float*)d_in[2];
  const float* sinT = (const float*)d_in[3];
  const float* Wq   = (const float*)d_in[4];
  const float* Wk   = (const float*)d_in[5];
  const float* Wv   = (const float*)d_in[6];
  const float* Wo   = (const float*)d_in[7];

  char* ws = (char*)d_ws;
  size_t off = 0;
  auto alloc = [&](size_t bytes) -> char* {
    char* p = ws + off;
    off += (bytes + 255) & ~(size_t)255;
    return p;
  };
  __bf16* xb   = (__bf16*)alloc((size_t)S_SEQ * D_IN * 2);
  __bf16* wqt  = (__bf16*)alloc((size_t)HDTOT * D_IN * 2);
  __bf16* wkt  = (__bf16*)alloc((size_t)GDTOT * D_IN * 2);
  __bf16* wvt  = (__bf16*)alloc((size_t)GDTOT * D_IN * 2);
  __bf16* wot  = (__bf16*)alloc((size_t)D_IN * HDTOT * 2);
  __bf16* qbuf = (__bf16*)alloc((size_t)NH * S_SEQ * DHEAD * 2);
  __bf16* kbuf = (__bf16*)alloc((size_t)NG * S_SEQ * DHEAD * 2);
  __bf16* vbuf = (__bf16*)alloc((size_t)NG * DHEAD * S_SEQ * 2);
  __bf16* ctx  = xb;  // x's bf16 buffer is dead after the projections

  dim3 B(256);
  k_cvt_f32_bf16<<<(S_SEQ * D_IN / 4 + 255) / 256, B, 0, stream>>>(x, xb,
                                                                   S_SEQ * D_IN / 4);
  k_transpose_cvt<<<(D_IN / 32) * (HDTOT / 32), B, 0, stream>>>(Wq, wqt, D_IN, HDTOT);
  k_transpose_cvt<<<(D_IN / 32) * (GDTOT / 32), B, 0, stream>>>(Wk, wkt, D_IN, GDTOT);
  k_transpose_cvt<<<(D_IN / 32) * (GDTOT / 32), B, 0, stream>>>(Wv, wvt, D_IN, GDTOT);
  k_transpose_cvt<<<(HDTOT / 32) * (D_IN / 32), B, 0, stream>>>(Wo, wot, HDTOT, D_IN);
  k_gemm<0><<<(128 * (HDTOT / 128)) / 8, B, 0, stream>>>(xb, wqt, cosT, sinT, qbuf,
                                                         D_IN, HDTOT / 128);
  k_gemm<0><<<(128 * (GDTOT / 128)) / 8, B, 0, stream>>>(xb, wkt, cosT, sinT, kbuf,
                                                         D_IN, GDTOT / 128);
  k_gemm<3><<<(128 * (GDTOT / 128)) / 8, B, 0, stream>>>(xb, wvt, nullptr, nullptr,
                                                         vbuf, D_IN, GDTOT / 128);
  k_attn<<<NH * (S_SEQ / 128), B, 0, stream>>>(qbuf, kbuf, vbuf, ctx);
  k_gemm<2><<<(128 * (HDTOT / 128)) / 8, B, 0, stream>>>(ctx, wot, nullptr, nullptr,
                                                         d_out, HDTOT, HDTOT / 128);
}
